// CausalSelfAttention_73297911873974
// MI455X (gfx1250) — matrix operand order repaired and checked by reading, not compiled, not run on hardware
//
#include <hip/hip_runtime.h>
#include <hip/hip_bf16.h>

typedef __attribute__((ext_vector_type(16))) _Float16 v16h;
typedef __attribute__((ext_vector_type(8)))  float    v8f;
typedef __attribute__((ext_vector_type(4)))  unsigned int v4u;
typedef __attribute__((ext_vector_type(4)))  int v4i;
typedef __attribute__((ext_vector_type(8)))  int v8i;

#define T_SEQ   4096
#define D_MODEL 1024
#define NH      16
#define HD      64
#define D3      3072

#if __has_builtin(__builtin_amdgcn_global_load_async_to_lds_b128)
#define HAS_ASYNC_LDS 1
#else
#define HAS_ASYNC_LDS 0
#endif

#if __has_builtin(__builtin_amdgcn_tensor_load_to_lds)
#define HAS_TDM 1
#else
#define HAS_TDM 0
#endif

typedef __attribute__((address_space(1))) v4i* gptr_v4i;
typedef __attribute__((address_space(3))) v4i* lptr_v4i;

union FragH { v16h v; v4u u[2]; };

__device__ __forceinline__ v16h load_frag_2(const _Float16* p0, const _Float16* p1) {
    FragH f;
    f.u[0] = *(const v4u*)p0;
    f.u[1] = *(const v4u*)p1;
    return f.v;
}

// 16-byte global -> LDS copy; async (no VGPR round-trip, ASYNCcnt) when available.
__device__ __forceinline__ void copy16_g2l(const _Float16* g, _Float16* l) {
#if HAS_ASYNC_LDS
    __builtin_amdgcn_global_load_async_to_lds_b128(
        (gptr_v4i)(void*)(_Float16*)g,   // global src (addrspace 1, int4*)
        (lptr_v4i)(void*)l,              // LDS dst    (addrspace 3, int4*)
        0, 0);
#else
    *(v4u*)l = *(const v4u*)g;
#endif
}

template <int N>
__device__ __forceinline__ void async_wait() {
#if HAS_ASYNC_LDS
#if __has_builtin(__builtin_amdgcn_s_wait_asynccnt)
    __builtin_amdgcn_s_wait_asynccnt(N);
#else
    asm volatile("s_wait_asynccnt %0" ::"i"(N));
#endif
#endif
}

#if HAS_TDM
__device__ __forceinline__ unsigned lds_offset_of(const void* p) {
    return (unsigned)(unsigned long long)
        (__attribute__((address_space(3))) const void*)p;
}

// Issue one TDM 2D tile load: tile_d1 rows x tile_d0 elements (2-byte elems),
// row stride stride_d0 elements, into LDS at lds_off with row padding:
// pad after 2^(pad_interval_enc+1) DWORDs, pad amount (pad_amount_enc+1) DWORDs.
__device__ __forceinline__ void tdm_load_2d(const _Float16* gaddr, unsigned lds_off,
                                            unsigned tensor_d0, unsigned tensor_d1,
                                            unsigned tile_d0, unsigned tile_d1,
                                            unsigned stride_d0,
                                            unsigned pad_en, unsigned pad_interval_enc,
                                            unsigned pad_amount_enc) {
    unsigned long long ga = (unsigned long long)gaddr;
    v4u g0;
    g0.x = 1u;                                            // count=1 (valid), user mode
    g0.y = lds_off;                                       // lds_addr (bytes)
    g0.z = (unsigned)(ga & 0xFFFFFFFFu);                  // global_addr[31:0]
    g0.w = (unsigned)((ga >> 32) & 0x01FFFFFFu) | (2u << 30);  // addr[56:32] | type=2
    v8i g1;
    g1[0] = (int)((1u << 16) |                            // data_size=1 (2 bytes)
                  (pad_en << 20) |
                  (pad_interval_enc << 22) |
                  (pad_amount_enc << 25));
    g1[1] = (int)((tensor_d0 & 0xFFFFu) << 16);           // tensor_dim0[15:0]
    g1[2] = (int)((tensor_d0 >> 16) | ((tensor_d1 & 0xFFFFu) << 16));
    g1[3] = (int)((tensor_d1 >> 16) | (tile_d0 << 16));   // tensor_dim1[31:16] | tile_dim0
    g1[4] = (int)(tile_d1 & 0xFFFFu);                     // tile_dim1 (tile_dim2 = 0)
    g1[5] = (int)stride_d0;                               // tensor_dim0_stride[31:0]
    g1[6] = 0;
    g1[7] = 0;
    v4i z4 = {0, 0, 0, 0};
    v8i z8 = {0, 0, 0, 0, 0, 0, 0, 0};
    __builtin_amdgcn_tensor_load_to_lds(g0, g1, z4, z4, z8, 0);
}
#endif

template <int N>
__device__ __forceinline__ void tensor_wait() {
#if HAS_TDM
    __builtin_amdgcn_s_wait_tensorcnt(N);
#endif
}

// ---------------------------------------------------------------------------
// Kernel 1: f32 -> f16 conversion (4-wide)
// ---------------------------------------------------------------------------
__global__ void cvt_f32_f16(const float4* __restrict__ in,
                            _Float16* __restrict__ out, int n4) {
    int i = blockIdx.x * blockDim.x + threadIdx.x;
    int stride = gridDim.x * blockDim.x;
    for (; i < n4; i += stride) {
        float4 f = in[i];
        out[4 * i + 0] = (_Float16)f.x;
        out[4 * i + 1] = (_Float16)f.y;
        out[4 * i + 2] = (_Float16)f.z;
        out[4 * i + 3] = (_Float16)f.w;
    }
}

// ---------------------------------------------------------------------------
// Kernels 2 & 4: C[M,N] = A[M,K] * B[N,K]^T  (row-major, K contiguous)
// 256 threads = 8 wave32. Block tile 128x128, K-step 32, double-buffered LDS.
// Tiles are streamed by the Tensor Data Mover (one descriptor per 128x32 tile,
// wave 0 issues, TENSORcnt tracks) with per-lane async-LDS loads as fallback.
// ---------------------------------------------------------------------------
template <bool OUT_HALF>
__global__ void __launch_bounds__(256)
gemm_wmma(const _Float16* __restrict__ A,
          const _Float16* __restrict__ B,
          void* __restrict__ Cout, int M, int N, int K) {
    __shared__ __align__(16) _Float16 lA[2][128][40];   // [buf][row][k], +8 pad
    __shared__ __align__(16) _Float16 lB[2][128][40];   // [buf][col][k], +8 pad

    const int tid   = threadIdx.x;
    const int lid   = tid & 31;
    const int wave  = tid >> 5;
    const int wm    = wave & 3;        // 32-row slice
    const int wn    = wave >> 2;       // 64-col slice
    const int hi    = lid >> 4;        // lane half (0/1)
    const int l16   = lid & 15;
    const int kBase = hi * 8;          // A-fragment K sub-offset per ISA layout

    const int blockM = blockIdx.y * 128;
    const int blockN = blockIdx.x * 128;

    v8f acc[2][4] = {};

    const int KT = K >> 5;

    auto stage = [&](int kt, int buf) {
#if HAS_TDM
        // One DMA per tile: 128 rows x 32 elems, row stride K elems; LDS rows
        // padded: 16 DWORDs (enc 3 = 2^4) then +4 DWORDs (enc 3) -> 40-half rows.
        if (wave == 0 && lid == 0) {
            tdm_load_2d(A + (size_t)blockM * K + kt * 32, lds_offset_of(&lA[buf][0][0]),
                        (unsigned)K, (unsigned)M, 32u, 128u, (unsigned)K, 1u, 3u, 3u);
            tdm_load_2d(B + (size_t)blockN * K + kt * 32, lds_offset_of(&lB[buf][0][0]),
                        (unsigned)K, (unsigned)N, 32u, 128u, (unsigned)K, 1u, 3u, 3u);
        }
#else
#pragma unroll
        for (int i = 0; i < 2; ++i) {
            int chunk = tid + i * 256;           // 0..511
            int r = chunk >> 2;                  // row 0..127
            int c = (chunk & 3) * 8;             // k offset 0/8/16/24
            copy16_g2l(A + (size_t)(blockM + r) * K + kt * 32 + c, &lA[buf][r][c]);
            copy16_g2l(B + (size_t)(blockN + r) * K + kt * 32 + c, &lB[buf][r][c]);
        }
#endif
    };

    stage(0, 0);

    for (int kt = 0; kt < KT; ++kt) {
        const int cur = kt & 1;
        if (kt + 1 < KT) {
            stage(kt + 1, cur ^ 1);   // stream next tile while we compute
#if HAS_TDM
            tensor_wait<2>();         // current tile's 2 DMAs retired
#else
            async_wait<4>();          // current tile (issued earlier) resident
#endif
        } else {
#if HAS_TDM
            tensor_wait<0>();
#else
            async_wait<0>();
#endif
        }
        __syncthreads();

        // Load ALL fragments first (one clause, one wait), then WMMAs back-to-back
        v16h a[2], b[4];
#pragma unroll
        for (int mt = 0; mt < 2; ++mt) {
            const _Float16* row = &lA[cur][wm * 32 + mt * 16 + l16][0];
            a[mt] = load_frag_2(row + kBase, row + 16 + kBase);
        }
#pragma unroll
        for (int nt = 0; nt < 4; ++nt) {
            const _Float16* row = &lB[cur][wn * 64 + nt * 16 + l16][0];
            b[nt] = load_frag_2(row + hi * 8, row + 16 + hi * 8);
        }
#pragma unroll
        for (int mt = 0; mt < 2; ++mt)
#pragma unroll
            for (int nt = 0; nt < 4; ++nt)
                acc[mt][nt] = __builtin_amdgcn_wmma_f32_16x16x32_f16(
                    false, a[mt], false, b[nt], (short)0, acc[mt][nt], false, false);
        __syncthreads();   // all waves done reading before this buffer is restaged
    }

    // Store C: lane holds c[j] = C[m = j + 8*hi][n = l16] per 16x16 tile
#pragma unroll
    for (int mt = 0; mt < 2; ++mt) {
#pragma unroll
        for (int nt = 0; nt < 4; ++nt) {
            int col = blockN + wn * 64 + nt * 16 + l16;
#pragma unroll
            for (int j = 0; j < 8; ++j) {
                int row = blockM + wm * 32 + mt * 16 + hi * 8 + j;
                float v = acc[mt][nt][j];
                if (OUT_HALF)
                    ((_Float16*)Cout)[(size_t)row * N + col] = (_Float16)v;
                else
                    ((float*)Cout)[(size_t)row * N + col] = v;
            }
        }
    }
}

// ---------------------------------------------------------------------------
// Kernel 3: causal flash attention.
// Grid: (T/64)*H blocks; block = 128 threads = 4 waves; wave owns 16 q rows.
// qkv layout: [T][3*D] f16; head h uses cols [h*64, h*64+64) of q/k/v slices.
// ---------------------------------------------------------------------------
__global__ void __launch_bounds__(128)
flash_attn(const _Float16* __restrict__ qkv, _Float16* __restrict__ y) {
    __shared__ __align__(16) _Float16 lK[64][72];       // [key][d]
    __shared__ __align__(16) _Float16 lV[64][72];       // transposed: [d][key]
    __shared__ __align__(16) _Float16 lP[4][16][72];    // per-wave P tile [m][key]

    const int tid   = threadIdx.x;
    const int lid   = tid & 31;
    const int wave  = tid >> 5;        // 0..3
    const int hi    = lid >> 4;
    const int l16   = lid & 15;
    const int kBase = hi * 8;

    const int h     = blockIdx.x & (NH - 1);
    const int qb    = blockIdx.x >> 4;       // query block (64 rows)
    const int qBase = qb * 64;
    const int qWave = qBase + wave * 16;

    const _Float16* qg = qkv + h * HD;
    const _Float16* kg = qkv + D_MODEL + h * HD;
    const _Float16* vg = qkv + 2 * D_MODEL + h * HD;

    // Preload this wave's Q fragments (16 rows x HD=64 -> 2 K-steps), kept in VGPRs
    v16h aQ[2];
    {
        const _Float16* row = qg + (size_t)(qWave + l16) * D3;
#pragma unroll
        for (int ks = 0; ks < 2; ++ks)
            aQ[ks] = load_frag_2(row + ks * 32 + kBase, row + ks * 32 + 16 + kBase);
    }

    float mrow[8], lrow[8];
    v8f o[4] = {};
#pragma unroll
    for (int r = 0; r < 8; ++r) { mrow[r] = -1e30f; lrow[r] = 0.0f; }

    const float scale = 0.125f;   // 1/sqrt(HD)

    for (int jb = 0; jb <= qb; ++jb) {       // causal: skip fully-masked key blocks
        const int keyBase = jb * 64;

        // Stage K [key][d] (async path) and V transposed [d][key] (VGPR path)
#pragma unroll
        for (int i = 0; i < 4; ++i) {
            int chunk = tid + i * 128;       // 0..511
            int r = chunk >> 3;              // key 0..63
            int c = (chunk & 7) * 8;         // d offset
            copy16_g2l(kg + (size_t)(keyBase + r) * D3 + c, &lK[r][c]);
            _Float16 tmp[8];
            *(v4u*)tmp = *(const v4u*)(vg + (size_t)(keyBase + r) * D3 + c);
#pragma unroll
            for (int e = 0; e < 8; ++e) lV[c + e][r] = tmp[e];
        }
        // Warm L2 for the next key block while we compute this one
        if (jb + 1 <= qb) {
            const _Float16* nk = kg + (size_t)(keyBase + 64 + (tid >> 1)) * D3 + (tid & 1) * 32;
            const _Float16* nv = vg + (size_t)(keyBase + 64 + (tid >> 1)) * D3 + (tid & 1) * 32;
            __builtin_prefetch(nk, 0, 0);
            __builtin_prefetch(nv, 0, 0);
        }
        async_wait<0>();
        __syncthreads();

        // S = Q * K^T : load all 8 B-fragments, then 8 WMMAs back-to-back
        v16h bK[4][2];
#pragma unroll
        for (int nt = 0; nt < 4; ++nt)
#pragma unroll
            for (int ks = 0; ks < 2; ++ks) {
                const _Float16* row = &lK[nt * 16 + l16][ks * 32 + hi * 8];
                bK[nt][ks] = load_frag_2(row, row + 16);
            }
        v8f s[4] = {};
#pragma unroll
        for (int nt = 0; nt < 4; ++nt)
#pragma unroll
            for (int ks = 0; ks < 2; ++ks)
                s[nt] = __builtin_amdgcn_wmma_f32_16x16x32_f16(
                    false, aQ[ks], false, bK[nt][ks], (short)0, s[nt], false, false);

        // Causal mask + online softmax (row r lives in one 16-lane half)
#pragma unroll
        for (int r = 0; r < 8; ++r) {
            int q = qWave + hi * 8 + r;
            float v0[4];
            float rmax = -1e30f;
#pragma unroll
            for (int nt = 0; nt < 4; ++nt) {
                int key = keyBase + nt * 16 + l16;
                float vv = s[nt][r] * scale;
                if (key > q) vv = -1e30f;
                v0[nt] = vv;
                rmax = fmaxf(rmax, vv);
            }
#pragma unroll
            for (int off = 8; off >= 1; off >>= 1)
                rmax = fmaxf(rmax, __shfl_xor(rmax, off, 32));
            float mnew  = fmaxf(mrow[r], rmax);
            float alpha = __expf(mrow[r] - mnew);
            float rsum  = 0.0f;
#pragma unroll
            for (int nt = 0; nt < 4; ++nt) {
                float p = __expf(v0[nt] - mnew);
                rsum += p;
                lP[wave][hi * 8 + r][nt * 16 + l16] = (_Float16)p;
            }
#pragma unroll
            for (int off = 8; off >= 1; off >>= 1)
                rsum += __shfl_xor(rsum, off, 32);
            lrow[r] = lrow[r] * alpha + rsum;
            mrow[r] = mnew;
#pragma unroll
            for (int nt = 0; nt < 4; ++nt)
                o[nt][r] *= alpha;
        }
        __syncthreads();

        // O += P * V : load all fragments, then 8 WMMAs back-to-back
        v16h aP[2], bV[4][2];
#pragma unroll
        for (int ks = 0; ks < 2; ++ks) {
            const _Float16* prow = &lP[wave][l16][ks * 32];
            aP[ks] = load_frag_2(prow + kBase, prow + 16 + kBase);
#pragma unroll
            for (int nt = 0; nt < 4; ++nt) {
                const _Float16* vrow = &lV[nt * 16 + l16][ks * 32 + hi * 8];
                bV[nt][ks] = load_frag_2(vrow, vrow + 16);
            }
        }
#pragma unroll
        for (int ks = 0; ks < 2; ++ks)
#pragma unroll
            for (int nt = 0; nt < 4; ++nt)
                o[nt] = __builtin_amdgcn_wmma_f32_16x16x32_f16(
                    false, aP[ks], false, bV[nt][ks], (short)0, o[nt], false, false);
        __syncthreads();
    }

    // y[t][h*64 + d] = O / l
#pragma unroll
    for (int nt = 0; nt < 4; ++nt) {
        int col = h * HD + nt * 16 + l16;
#pragma unroll
        for (int r = 0; r < 8; ++r) {
            int row = qWave + hi * 8 + r;
            y[(size_t)row * D_MODEL + col] = (_Float16)(o[nt][r] / lrow[r]);
        }
    }
}

// ---------------------------------------------------------------------------
extern "C" void kernel_launch(void* const* d_in, const int* in_sizes, int n_in,
                              void* d_out, int out_size, void* d_ws, size_t ws_size,
                              hipStream_t stream) {
    const float* x      = (const float*)d_in[0];
    const float* w_attn = (const float*)d_in[1];
    const float* w_proj = (const float*)d_in[2];
    float* out = (float*)d_out;

    char* ws = (char*)d_ws;
    _Float16* xh   = (_Float16*)ws; ws += (size_t)T_SEQ * D_MODEL * 2;
    _Float16* wah  = (_Float16*)ws; ws += (size_t)D3 * D_MODEL * 2;
    _Float16* wph  = (_Float16*)ws; ws += (size_t)D_MODEL * D_MODEL * 2;
    _Float16* qkvh = (_Float16*)ws; ws += (size_t)T_SEQ * D3 * 2;
    _Float16* yh   = (_Float16*)ws;

    cvt_f32_f16<<<256, 256, 0, stream>>>((const float4*)x, xh, T_SEQ * D_MODEL / 4);
    cvt_f32_f16<<<256, 256, 0, stream>>>((const float4*)w_attn, wah, D3 * D_MODEL / 4);
    cvt_f32_f16<<<64, 256, 0, stream>>>((const float4*)w_proj, wph, D_MODEL * D_MODEL / 4);

    gemm_wmma<true><<<dim3(D3 / 128, T_SEQ / 128), 256, 0, stream>>>(
        xh, wah, (void*)qkvh, T_SEQ, D3, D_MODEL);

    flash_attn<<<(T_SEQ / 64) * NH, 128, 0, stream>>>(qkvh, yh);

    gemm_wmma<false><<<dim3(D_MODEL / 128, T_SEQ / 128), 256, 0, stream>>>(
        yh, wph, (void*)out, T_SEQ, D_MODEL, D_MODEL);
}
